// Model_23055384445634
// MI455X (gfx1250) — compile-verified, matrix operand order checked by reading, not run on hardware
//
#include <hip/hip_runtime.h>

typedef __attribute__((ext_vector_type(16))) __bf16 v16bf;
typedef __attribute__((ext_vector_type(8)))  float  v8f;

#define B_   4096
#define NI_  784
#define NIP  800      // K padded to multiple of 32
#define NH_  1000
#define NHP  1008     // N padded to multiple of 16 (fc1) 
#define NHK  1024     // K padded to multiple of 32 (fc2)
#define NO_  10
#define TSTEPS 25
#define BETA_ 0.9f
#define THR_  1.0f

// ---------------- conversion kernels (f32 -> bf16, zero padded) --------------

__global__ __launch_bounds__(256) void snn_cvt_x(const float* __restrict__ x,
                                                 __bf16* __restrict__ xb) {
    int idx = blockIdx.x * 256 + threadIdx.x;          // over B_*NIP
    if (idx >= B_ * NIP) return;
    int r = idx / NIP, k = idx - r * NIP;
    float v = (k < NI_) ? x[r * NI_ + k] : 0.0f;
    xb[idx] = (__bf16)v;
}

__global__ __launch_bounds__(256) void snn_cvt_w1(const float* __restrict__ w1,
                                                  __bf16* __restrict__ wb) {
    int idx = blockIdx.x * 256 + threadIdx.x;          // over NHP*NIP
    if (idx >= NHP * NIP) return;
    int r = idx / NIP, k = idx - r * NIP;
    float v = (r < NH_ && k < NI_) ? w1[r * NI_ + k] : 0.0f;
    wb[idx] = (__bf16)v;
}

// ---------------- fc1: cur1 = x @ w1^T + b1  (bf16 WMMA, f32 accum) ----------
// one wave per 16x16 output tile; K = 800 -> 25 chained v_wmma_f32_16x16x32_bf16

__global__ __launch_bounds__(32) void snn_fc1_wmma(const __bf16* __restrict__ xb,
                                                   const __bf16* __restrict__ w1b,
                                                   const float* __restrict__ b1,
                                                   float* __restrict__ cur1) {
    const int l    = threadIdx.x;        // 0..31
    const int i16  = l & 15;
    const int hh   = l >> 4;             // lane half selects K sub-chunk
    const int n0   = blockIdx.x * 16;    // 63 tiles -> 0..1007
    const int m0   = blockIdx.y * 16;    // 256 tiles -> 0..4095

    const __bf16* arow = xb  + (size_t)(m0 + i16) * NIP;  // A: row of x
    const __bf16* brow = w1b + (size_t)(n0 + i16) * NIP;  // B: row of w1 (K-contig)

    v8f acc = {};
#pragma unroll
    for (int kc = 0; kc < NIP / 32; ++kc) {
        const int k0 = kc * 32 + hh * 8;
        v16bf A, Bv;
        *((uint4*)&A)      = *(const uint4*)(arow + k0);
        *((uint4*)&A + 1)  = *(const uint4*)(arow + k0 + 16);
        *((uint4*)&Bv)     = *(const uint4*)(brow + k0);
        *((uint4*)&Bv + 1) = *(const uint4*)(brow + k0 + 16);
        acc = __builtin_amdgcn_wmma_f32_16x16x32_bf16(false, A, false, Bv,
                                                      (short)0, acc, false, false);
    }

    const int n = n0 + i16;
    if (n < NH_) {
        const float bn = b1[n];
#pragma unroll
        for (int r = 0; r < 8; ++r) {
            const int m = m0 + r + hh * 8;   // C layout: VGPR r -> M=r (+8 for hi half)
            cur1[(size_t)m * NH_ + n] = acc[r] + bn;
        }
    }
}

// ---------------- timestep loop: LIF + fc2 WMMA per step ---------------------
// block = 16 batch rows. mem1 in registers, spikes + w2 in LDS, mem2 in the
// WMMA f32 accumulator layout of wave 0.

__global__ __launch_bounds__(256) void snn_steps(const float* __restrict__ cur1,
                                                 const float* __restrict__ w2,
                                                 const float* __restrict__ b2,
                                                 float* __restrict__ out) {
    __shared__ __bf16 spk1[16][NHK];   // 32 KB
    __shared__ __bf16 w2s[16][NHK];    // 32 KB

    const int tid  = threadIdx.x;
    const int tile = blockIdx.x;       // 0..255 -> batch rows tile*16 .. +15
    const int lane = tid & 31;
    const int wave = tid >> 5;
    const int i16  = lane & 15;
    const int hh   = lane >> 4;

    // stage w2 (bf16, zero-padded to 16x1024) and zero spike padding
    for (int idx = tid; idx < 16 * NHK; idx += 256) {
        const int n = idx >> 10, k = idx & (NHK - 1);
        float v = (n < NO_ && k < NH_) ? w2[n * NH_ + k] : 0.0f;
        w2s[n][k]  = (__bf16)v;
        spk1[n][k] = (__bf16)0.0f;
    }
    __syncthreads();

    // per-thread ownership: row = tid/16, cols = (tid%16) + 16*j, j<63
    const int row  = tid >> 4;
    const int cbase = tid & 15;
    const float* crow = cur1 + (size_t)(tile * 16 + row) * NH_;

    float m1[63], c1[63];
#pragma unroll
    for (int j = 0; j < 63; ++j) {
        const int col = cbase + (j << 4);
        m1[j] = 0.0f;
        c1[j] = (col < NH_) ? crow[col] : 0.0f;
    }

    v8f mem2 = {};
    float b2n = 0.0f;
    if (wave == 0 && i16 < NO_) b2n = b2[i16];

    for (int s = 0; s < TSTEPS; ++s) {
        // LIF layer 1: reset-by-subtraction from previous mem, then spike
#pragma unroll
        for (int j = 0; j < 63; ++j) {
            const int col = cbase + (j << 4);
            if (col < NH_) {
                const float prev = m1[j];
                const float rst  = (prev > THR_) ? THR_ : 0.0f;
                const float m    = BETA_ * prev + c1[j] - rst;
                m1[j] = m;
                spk1[row][col] = (__bf16)((m > THR_) ? 1.0f : 0.0f);
            }
        }
        __syncthreads();

        if (wave == 0) {   // wave-uniform branch: EXEC all ones for WMMA
            const __bf16* arow = &spk1[i16][0];
            const __bf16* brow = &w2s[i16][0];
            v8f acc = {};
#pragma unroll
            for (int kc = 0; kc < NHK / 32; ++kc) {
                const int k0 = kc * 32 + hh * 8;
                v16bf A, Bv;
                *((uint4*)&A)      = *(const uint4*)(arow + k0);
                *((uint4*)&A + 1)  = *(const uint4*)(arow + k0 + 16);
                *((uint4*)&Bv)     = *(const uint4*)(brow + k0);
                *((uint4*)&Bv + 1) = *(const uint4*)(brow + k0 + 16);
                acc = __builtin_amdgcn_wmma_f32_16x16x32_bf16(false, A, false, Bv,
                                                              (short)0, acc, false, false);
            }
            // LIF layer 2 on the accumulator layout; emit spikes
#pragma unroll
            for (int r = 0; r < 8; ++r) {
                const float cur2 = acc[r] + b2n;
                const float prev = mem2[r];
                const float rst  = (prev > THR_) ? THR_ : 0.0f;
                const float m    = BETA_ * prev + cur2 - rst;
                mem2[r] = m;
                if (i16 < NO_) {
                    const int mrow = tile * 16 + r + hh * 8;
                    out[(size_t)s * (B_ * NO_) + (size_t)mrow * NO_ + i16] =
                        (m > THR_) ? 1.0f : 0.0f;
                }
            }
        }
        __syncthreads();   // spk1 consumed before next step overwrites it
    }
}

// ---------------------------------------------------------------------------

extern "C" void kernel_launch(void* const* d_in, const int* in_sizes, int n_in,
                              void* d_out, int out_size, void* d_ws, size_t ws_size,
                              hipStream_t stream) {
    const float* x  = (const float*)d_in[0];   // [4096,784]
    const float* w1 = (const float*)d_in[1];   // [1000,784]
    const float* b1 = (const float*)d_in[2];   // [1000]
    const float* w2 = (const float*)d_in[3];   // [10,1000]
    const float* b2 = (const float*)d_in[4];   // [10]
    float* out = (float*)d_out;                // [25,4096,10]

    char* ws = (char*)d_ws;
    __bf16* xb   = (__bf16*)ws;                               // 4096*800*2  = 6,553,600 B
    __bf16* w1b  = (__bf16*)(ws + (size_t)B_ * NIP * 2);      // 1008*800*2  = 1,612,800 B
    float*  cur1 = (float*)(ws + (size_t)B_ * NIP * 2
                               + (size_t)NHP * NIP * 2);      // 4096*1000*4 = 16,384,000 B

    snn_cvt_x <<<(B_ * NIP + 255) / 256, 256, 0, stream>>>(x, xb);
    snn_cvt_w1<<<(NHP * NIP + 255) / 256, 256, 0, stream>>>(w1, w1b);

    dim3 g1(NHP / 16, B_ / 16);   // (63, 256) tiles, one wave each
    snn_fc1_wmma<<<g1, 32, 0, stream>>>(xb, w1b, b1, cur1);

    snn_steps<<<B_ / 16, 256, 0, stream>>>(cur1, w2, b2, out);
}
